// AttentionBandLayer_19215683682902
// MI455X (gfx1250) — matrix operand order repaired and checked by reading, not compiled, not run on hardware
//
#include <hip/hip_runtime.h>
#include <math.h>

// ---------------------------------------------------------------------------
// CDNA5 (gfx1250) transformer layer: bf16 WMMA GEMMs with TDM (tensor DMA)
// double-buffered LDS staging + flash banded attention.
// All WMMA fragments are contiguous 32B/lane -> 2x b128 loads.
// ---------------------------------------------------------------------------

typedef __attribute__((ext_vector_type(16))) __bf16          v16bf;
typedef __attribute__((ext_vector_type(8)))  float           v8f;
typedef __attribute__((ext_vector_type(16))) unsigned short  v16us;
typedef __attribute__((ext_vector_type(4)))  unsigned int    v4u;
typedef __attribute__((ext_vector_type(8)))  int             v8i_t;
typedef __attribute__((ext_vector_type(4)))  int             v4i_t;

#define TDM_OK __has_builtin(__builtin_amdgcn_tensor_load_to_lds)

#define LDS_PITCH 40   // 32 bf16 (16 dw) + 4 dw pad: 16B-aligned, conflict-free

union FragU { v16us u; v16bf b; };

__device__ __forceinline__ unsigned short f2bf(float f) {
  unsigned int u = __float_as_uint(f);
  u += 0x7FFFu + ((u >> 16) & 1u);            // round-to-nearest-even
  return (unsigned short)(u >> 16);
}
__device__ __forceinline__ float bf2f(unsigned short h) {
  return __uint_as_float(((unsigned int)h) << 16);
}

__device__ __forceinline__ v8f wmma_bf16(v16us a, v16us b, v8f c) {
  FragU ua, ub; ua.u = a; ub.u = b;
  return __builtin_amdgcn_wmma_f32_16x16x32_bf16(
      false, ua.b, false, ub.b, (short)0, c, false, false);
}

// ---- Fragment loaders (contiguous-K order, same for A and B, wave32) -------------------------
// A 16x32 bf16: lane m=lane&15 holds row m; half-wave selects K {0-15}
// vs {16-31}.  Two 16B vector loads per lane.
__device__ __forceinline__ v16us load_frag_a(const unsigned short* base,
                                             int lda, int lane) {
  int m = lane & 15;
  int kg = (lane >> 4) << 4;                  // 0 or 16
  const unsigned short* p = base + (long)m * lda + kg;
  union { v16us v; v4u q[2]; } t;
  t.q[0] = *(const v4u*)p;
  t.q[1] = *(const v4u*)(p + 8);
  return t.v;
}

// B 32x16 from column-contiguous storage (each column's 32 K-values are
// contiguous with pitch ld): one 32B contiguous read per lane.
__device__ __forceinline__ v16us load_frag_b_cc(const unsigned short* base,
                                                long ld, int lane) {
  int n = lane & 15;
  int kg = (lane >> 4) << 4;                  // 0 or 16
  const unsigned short* p = base + (long)n * ld + kg;
  union { v16us v; v4u q[2]; } t;
  t.q[0] = ((const v4u*)p)[0];
  t.q[1] = ((const v4u*)p)[1];
  return t.v;
}

__device__ __forceinline__ float redmax16(float v) {
#pragma unroll
  for (int m = 1; m < 16; m <<= 1) v = fmaxf(v, __shfl_xor(v, m, 32));
  return v;
}
__device__ __forceinline__ float redsum16(float v) {
#pragma unroll
  for (int m = 1; m < 16; m <<= 1) v += __shfl_xor(v, m, 32);
  return v;
}

// ---------------------------------------------------------------------------
// Tensor Data Mover: 2D tile load global->LDS (ISA ch.8 D# descriptor).
// pad_interval=3 (16 dw) / pad_amount=3 (4 dw) reproduces the 40-us LDS pitch.
// ---------------------------------------------------------------------------
#if TDM_OK
__device__ __forceinline__ void tdm_load_tile(
    unsigned lds_addr, const void* gptr,
    unsigned tensor_d0, unsigned tensor_d1, unsigned long long stride0) {
  unsigned long long ga = (unsigned long long)gptr;
  v4u g0;
  g0.x = 1u;                                        // count=1 (valid user D#)
  g0.y = lds_addr;                                  // LDS byte address
  g0.z = (unsigned)(ga & 0xFFFFFFFFu);              // global_addr[31:0]
  g0.w = (unsigned)((ga >> 32) & 0x01FFFFFFu)       // global_addr[56:32]
       | 0x80000000u;                               // type=2 ("image")
  // group1: data_size=1 (2B), pad_enable, tile 32x128, pads (3,3)
  unsigned d0 = (1u << 16) | (1u << 20) | (3u << 22) | (3u << 25);
  unsigned d1 = (tensor_d0 & 0xFFFFu) << 16;
  unsigned d2 = (tensor_d0 >> 16) | ((tensor_d1 & 0xFFFFu) << 16);
  unsigned d3 = (tensor_d1 >> 16) | (32u << 16);    // tile_dim0 = 32
  unsigned d4 = 128u;                               // tile_dim1 = 128
  unsigned d5 = (unsigned)(stride0 & 0xFFFFFFFFull);
  unsigned d6 = (unsigned)((stride0 >> 32) & 0xFFFFull);
  v8i_t g1 = { (int)d0, (int)d1, (int)d2, (int)d3,
               (int)d4, (int)d5, (int)d6, 0 };
  v4i_t z4 = { 0, 0, 0, 0 };
#if defined(__clang_major__) && __clang_major__ >= 23
  v8i_t z8 = { 0, 0, 0, 0, 0, 0, 0, 0 };
  __builtin_amdgcn_tensor_load_to_lds(g0, g1, z4, z4, z8, 0);
#else
  __builtin_amdgcn_tensor_load_to_lds(g0, g1, z4, z4, 0);
#endif
}
#endif

// ---------------------------------------------------------------------------
// fp32 -> bf16 transposing conversion (weights): out[n*K+k] = in[k*N+n]
// ---------------------------------------------------------------------------
__global__ __launch_bounds__(256) void transpose_cvt_kernel(
    const float* __restrict__ in, unsigned short* __restrict__ out,
    int K, int N) {
  __shared__ unsigned short tile[32][33];
  int kb = blockIdx.x * 32, nb = blockIdx.y * 32;
  int tx = threadIdx.x & 31, ty = threadIdx.x >> 5;   // ty: 0..7
#pragma unroll
  for (int i = 0; i < 32; i += 8)
    tile[ty + i][tx] = f2bf(in[(long)(kb + ty + i) * N + nb + tx]);
  __syncthreads();
#pragma unroll
  for (int i = 0; i < 32; i += 8)
    out[(long)(nb + ty + i) * K + kb + tx] = tile[tx][ty + i];
}

// ---------------------------------------------------------------------------
// LayerNorm: one block per row of D=1024, writes bf16
// ---------------------------------------------------------------------------
__global__ __launch_bounds__(256) void ln_kernel(const float* __restrict__ x,
                                                 const float* __restrict__ g,
                                                 const float* __restrict__ b,
                                                 unsigned short* __restrict__ out,
                                                 int D) {
  __shared__ float sh[20];
  long row = blockIdx.x;
  const float* xr = x + row * D;
  float s = 0.f, s2 = 0.f;
  for (int i = threadIdx.x; i < D; i += blockDim.x) {
    float v = xr[i]; s += v; s2 += v * v;
  }
#pragma unroll
  for (int off = 16; off > 0; off >>= 1) {
    s  += __shfl_down(s, off, 32);
    s2 += __shfl_down(s2, off, 32);
  }
  int wid = threadIdx.x >> 5, lid = threadIdx.x & 31;
  if (lid == 0) { sh[wid] = s; sh[8 + wid] = s2; }
  __syncthreads();
  if (threadIdx.x == 0) {
    float a = 0.f, a2 = 0.f;
    for (int i = 0; i < 8; ++i) { a += sh[i]; a2 += sh[8 + i]; }
    sh[16] = a; sh[17] = a2;
  }
  __syncthreads();
  float mean = sh[16] / D;
  float var  = sh[17] / D - mean * mean;
  float inv  = rsqrtf(var + 1e-5f);
  for (int i = threadIdx.x; i < D; i += blockDim.x) {
    float v = (xr[i] - mean) * inv * g[i] + b[i];
    out[row * D + i] = f2bf(v);
  }
}

// ---------------------------------------------------------------------------
// Tiled bf16 WMMA GEMM: C[M,N] = A[M,K] x WT[N,K]^T + bias.
// Block tile 128x128, 8 waves of 32x64 (2x4 frags), K-step 32.
// A and B tiles are the SAME shape (128 rows x 32 contiguous K, pitch 40),
// both fed by TDM double-buffering.
// mode 0: bf16 out; mode 1: fp32 out = acc+bias+resid; mode 2: GELU bf16;
// mode 3: bf16 out transposed (outU[col*M+row]).
// ---------------------------------------------------------------------------
__global__ __launch_bounds__(256) void gemm_bf16_kernel(
    const unsigned short* __restrict__ A, const unsigned short* __restrict__ WT,
    const float* __restrict__ bias, const float* __restrict__ resid,
    float* __restrict__ outF, unsigned short* __restrict__ outU,
    int M, int N, int K, int mode) {
  __shared__ __align__(16) unsigned short As[2][128][LDS_PITCH];
  __shared__ __align__(16) unsigned short Bs[2][128][LDS_PITCH];

  const int t    = threadIdx.x;
  const int lane = t & 31;
  const int w    = t >> 5;
  const int wm   = w & 3;                  // 4 waves along M (32 rows each)
  const int wn   = w >> 2;                 // 2 waves along N (64 cols each)
  const int rowBase = blockIdx.y * 128;
  const int colBase = blockIdx.x * 128;
  const int nTiles  = K >> 5;

  v8f acc[2][4] = {};

#if TDM_OK
  if (w == 0) {
    tdm_load_tile((unsigned)(size_t)&As[0][0][0], A + (long)rowBase * K,
                  (unsigned)K, (unsigned)M, (unsigned long long)K);
    tdm_load_tile((unsigned)(size_t)&Bs[0][0][0], WT + (long)colBase * K,
                  (unsigned)K, (unsigned)N, (unsigned long long)K);
  }
#endif

  for (int it = 0; it < nTiles; ++it) {
    const int cur = it & 1;
    const bool have_next = (it + 1) < nTiles;

#if TDM_OK
    if (w == 0) {
      if (have_next) {
        int k1 = (it + 1) << 5;
        tdm_load_tile((unsigned)(size_t)&As[1 - cur][0][0],
                      A + (long)rowBase * K + k1,
                      (unsigned)K, (unsigned)M, (unsigned long long)K);
        tdm_load_tile((unsigned)(size_t)&Bs[1 - cur][0][0],
                      WT + (long)colBase * K + k1,
                      (unsigned)K, (unsigned)N, (unsigned long long)K);
        __builtin_amdgcn_s_wait_tensorcnt((short)2);  // current tile landed
      } else {
        __builtin_amdgcn_s_wait_tensorcnt((short)0);
      }
    }
    __syncthreads();
#else
    // Fallback staging: batch the 16 global dword loads, then store to LDS.
    {
      const int k0 = it << 5;
      unsigned int ra[8], rb[8];
#pragma unroll
      for (int i = 0; i < 8; ++i) {
        int e = t + i * 256;
        ra[i] = *(const unsigned int*)(A  + (long)(rowBase + (e >> 4)) * K + k0 + ((e & 15) << 1));
        rb[i] = *(const unsigned int*)(WT + (long)(colBase + (e >> 4)) * K + k0 + ((e & 15) << 1));
      }
#pragma unroll
      for (int i = 0; i < 8; ++i) {
        int e = t + i * 256;
        *(unsigned int*)(&As[cur][e >> 4][(e & 15) << 1]) = ra[i];
        *(unsigned int*)(&Bs[cur][e >> 4][(e & 15) << 1]) = rb[i];
      }
    }
    __syncthreads();
#endif

    v16us a0 = load_frag_a(&As[cur][wm * 32][0], LDS_PITCH, lane);
    v16us a1 = load_frag_a(&As[cur][wm * 32 + 16][0], LDS_PITCH, lane);
#pragma unroll
    for (int ni = 0; ni < 4; ++ni) {
      v16us bf = load_frag_b_cc(&Bs[cur][wn * 64 + ni * 16][0], LDS_PITCH, lane);
      acc[0][ni] = wmma_bf16(a0, bf, acc[0][ni]);
      acc[1][ni] = wmma_bf16(a1, bf, acc[1][ni]);
    }
    __syncthreads();   // everyone done with `cur` before it is overwritten
  }

  // --- epilogue ---
  const int n  = lane & 15;
  const int hi = lane >> 4;
#pragma unroll
  for (int mi = 0; mi < 2; ++mi)
#pragma unroll
    for (int ni = 0; ni < 4; ++ni) {
      int col = colBase + wn * 64 + ni * 16 + n;
      float bcol = bias[col];
#pragma unroll
      for (int r = 0; r < 8; ++r) {
        int row = rowBase + wm * 32 + mi * 16 + hi * 8 + r;
        float v = acc[mi][ni][r] + bcol;
        if (mode == 0) {
          outU[(long)row * N + col] = f2bf(v);
        } else if (mode == 1) {
          long off = (long)row * N + col;
          outF[off] = v + resid[off];
        } else if (mode == 2) {
          float gel = 0.5f * v * (1.0f + erff(v * 0.70710678118f));
          outU[(long)row * N + col] = f2bf(gel);
        } else {                                   // transposed bf16 (V^T)
          outU[(long)col * M + row] = f2bf(v);
        }
      }
    }
}

// ---------------------------------------------------------------------------
// RoPE in-place on q and k (bf16, [B*S, 1024], head dim 64)
// ---------------------------------------------------------------------------
__global__ void rope_kernel(unsigned short* __restrict__ q,
                            unsigned short* __restrict__ k,
                            long BS, int S) {
  long idx = (long)blockIdx.x * blockDim.x + threadIdx.x;
  long total = BS * 16 * 32;
  if (idx >= total) return;
  int  tt  = (int)(idx & 31);
  long tmp = idx >> 5;
  int  h   = (int)(tmp & 15);
  long row = tmp >> 4;
  int  pos = (int)(row % S);
  float freq = __expf(-0.2878231366f * (float)tt);   // 10000^(-2t/64)
  float ang  = (float)pos * freq;
  float c = __cosf(ang), sn = __sinf(ang);
  long base = row * 1024 + h * 64 + tt;
  float x1 = bf2f(q[base]), x2 = bf2f(q[base + 32]);
  q[base]      = f2bf(x1 * c - x2 * sn);
  q[base + 32] = f2bf(x2 * c + x1 * sn);
  x1 = bf2f(k[base]); x2 = bf2f(k[base + 32]);
  k[base]      = f2bf(x1 * c - x2 * sn);
  k[base + 32] = f2bf(x2 * c + x1 * sn);
}

// ---------------------------------------------------------------------------
// Flash-style banded+global attention.  1 wave per 16-query tile per (b,h).
// V is pre-transposed ([Dm][B*S]) so the PV B-operand is contiguous.
// ---------------------------------------------------------------------------
__global__ __launch_bounds__(32) void attn_kernel(
    const unsigned short* __restrict__ qb, const unsigned short* __restrict__ kbuf,
    const unsigned short* __restrict__ vT, const int* __restrict__ gidx,
    int n_glob, unsigned short* __restrict__ ctx, int S, int H, int BS) {
  __shared__ unsigned int sh_glob[64];                    // S=2048 bit-set
  __shared__ __align__(16) unsigned short sh_p[16][32];   // P tile (bf16)

  const int lane = threadIdx.x;
  const int qt0  = blockIdx.x * 16;
  const int bh   = blockIdx.y;
  const int b    = bh / H;
  const int h    = bh % H;
  const int hcol = h * 64;

  for (int i = lane; i < 64; i += 32) sh_glob[i] = 0u;
  __syncthreads();
  if (lane < n_glob) {
    int j = gidx[lane];
    atomicOr(&sh_glob[j >> 5], 1u << (j & 31));
  }
  __syncthreads();

  const bool has_gq = ((sh_glob[qt0 >> 5] >> (qt0 & 31)) & 0xFFFFu) != 0u;

  const long qrow = (long)b * S + qt0;
  v16us qa0 = load_frag_a(qb + qrow * 1024 + hcol,      1024, lane);
  v16us qa1 = load_frag_a(qb + qrow * 1024 + hcol + 32, 1024, lane);

  float mrun[8], lrun[8];
  v8f acc[4] = {};
#pragma unroll
  for (int r = 0; r < 8; ++r) { mrun[r] = -3e38f; lrun[r] = 0.f; }

  const int n  = lane & 15;
  const int hi = lane >> 4;
  const int qhi = qt0 + 15;

  for (int kb = 0; kb <= qhi; kb += 32) {
    bool need = (kb + 31 >= qt0 - 511) || has_gq || (sh_glob[kb >> 5] != 0u);
    if (!need) continue;

    long krow = (long)b * S + kb;
    v16us kt00 = load_frag_b_cc(kbuf + krow * 1024 + hcol,              1024, lane);
    v16us kt01 = load_frag_b_cc(kbuf + krow * 1024 + hcol + 32,         1024, lane);
    v16us kt10 = load_frag_b_cc(kbuf + (krow + 16) * 1024 + hcol,       1024, lane);
    v16us kt11 = load_frag_b_cc(kbuf + (krow + 16) * 1024 + hcol + 32,  1024, lane);

    v8f s0 = {}; s0 = wmma_bf16(qa0, kt00, s0); s0 = wmma_bf16(qa1, kt01, s0);
    v8f s1 = {}; s1 = wmma_bf16(qa0, kt10, s1); s1 = wmma_bf16(qa1, kt11, s1);

    const int j0 = kb + n, j1 = j0 + 16;
    const bool gj0 = (sh_glob[j0 >> 5] >> (j0 & 31)) & 1u;
    const bool gj1 = (sh_glob[j1 >> 5] >> (j1 & 31)) & 1u;

    float p0[8], p1[8];
#pragma unroll
    for (int r = 0; r < 8; ++r) {
      int i = qt0 + hi * 8 + r;
      bool gi = (sh_glob[i >> 5] >> (i & 31)) & 1u;
      bool a0 = (j0 <= i) && ((i - j0 < 512) || gj0 || gi);
      bool a1 = (j1 <= i) && ((i - j1 < 512) || gj1 || gi);
      float v0 = a0 ? s0[r] * 0.125f : -3e38f;
      float v1 = a1 ? s1[r] * 0.125f : -3e38f;
      float mrow = redmax16(fmaxf(v0, v1));
      float mnew = fmaxf(mrun[r], mrow);
      float alpha = __expf(mrun[r] - mnew);
      float e0 = a0 ? __expf(v0 - mnew) : 0.f;
      float e1 = a1 ? __expf(v1 - mnew) : 0.f;
      lrun[r] = lrun[r] * alpha + redsum16(e0 + e1);
      mrun[r] = mnew;
      p0[r] = e0; p1[r] = e1;
#pragma unroll
      for (int t2 = 0; t2 < 4; ++t2) acc[t2][r] *= alpha;
    }

    __syncthreads();
#pragma unroll
    for (int r = 0; r < 8; ++r) {
      sh_p[hi * 8 + r][n]      = f2bf(p0[r]);
      sh_p[hi * 8 + r][16 + n] = f2bf(p1[r]);
    }
    __syncthreads();
    v16us pa = load_frag_a(&sh_p[0][0], 32, lane);
#pragma unroll
    for (int ni = 0; ni < 4; ++ni) {
      const unsigned short* vcol = vT + (long)(hcol + ni * 16) * BS + krow;
      v16us bv = load_frag_b_cc(vcol, BS, lane);
      acc[ni] = wmma_bf16(pa, bv, acc[ni]);
    }
  }

#pragma unroll
  for (int ni = 0; ni < 4; ++ni)
#pragma unroll
    for (int r = 0; r < 8; ++r) {
      int i = qt0 + hi * 8 + r;
      float o = acc[ni][r] / lrun[r];
      ctx[((long)b * S + i) * 1024 + hcol + ni * 16 + n] = f2bf(o);
    }
}

// ---------------------------------------------------------------------------
extern "C" void kernel_launch(void* const* d_in, const int* in_sizes, int n_in,
                              void* d_out, int out_size, void* d_ws, size_t ws_size,
                              hipStream_t stream) {
  const int Bb = 2, S = 2048, Dm = 1024, Dff = 4096, H = 16;
  const long BS = (long)Bb * S;

  const float* hidden = (const float*)d_in[0];
  const float* ln1_g  = (const float*)d_in[1];
  const float* ln1_b  = (const float*)d_in[2];
  const float* wq = (const float*)d_in[3];  const float* bq = (const float*)d_in[4];
  const float* wk = (const float*)d_in[5];  const float* bk = (const float*)d_in[6];
  const float* wv = (const float*)d_in[7];  const float* bv = (const float*)d_in[8];
  const float* wo = (const float*)d_in[9];  const float* bo = (const float*)d_in[10];
  const float* ln2_g = (const float*)d_in[11];
  const float* ln2_b = (const float*)d_in[12];
  const float* w1 = (const float*)d_in[13]; const float* b1 = (const float*)d_in[14];
  const float* w2 = (const float*)d_in[15]; const float* b2 = (const float*)d_in[16];
  const int* gidx = (const int*)d_in[17];
  int n_glob = in_sizes[17]; if (n_glob > 32) n_glob = 32;
  float* out = (float*)d_out;

  char* ws = (char*)d_ws;
  size_t off = 0;
  auto alloc = [&](size_t bytes) -> char* {
    char* p = ws + off;
    off += (bytes + 255) & ~(size_t)255;
    return p;
  };
  unsigned short* xln  = (unsigned short*)alloc(BS * Dm * 2);
  unsigned short* wqT  = (unsigned short*)alloc((size_t)Dm * Dm * 2);
  unsigned short* wkT  = (unsigned short*)alloc((size_t)Dm * Dm * 2);
  unsigned short* wvT  = (unsigned short*)alloc((size_t)Dm * Dm * 2);
  unsigned short* woT  = (unsigned short*)alloc((size_t)Dm * Dm * 2);
  unsigned short* w1T  = (unsigned short*)alloc((size_t)Dm * Dff * 2);
  unsigned short* w2T  = (unsigned short*)alloc((size_t)Dff * Dm * 2);
  unsigned short* qbuf = (unsigned short*)alloc(BS * Dm * 2);
  unsigned short* kbuf = (unsigned short*)alloc(BS * Dm * 2);
  unsigned short* vTb  = (unsigned short*)alloc(BS * Dm * 2);   // [Dm][B*S]
  unsigned short* ctx  = (unsigned short*)alloc(BS * Dm * 2);
  float*          hbuf = (float*)alloc(BS * Dm * 4);
  unsigned short* ybuf = (unsigned short*)alloc(BS * Dm * 2);
  unsigned short* act = qbuf;   // BSxDff bf16 aliases dead q/k/v/ctx region

  auto tcvt = [&](const float* src, unsigned short* dst, int K, int N) {
    transpose_cvt_kernel<<<dim3(K / 32, N / 32), dim3(256), 0, stream>>>(src, dst, K, N);
  };
  auto gemm = [&](const unsigned short* A, const unsigned short* WT,
                  const float* bias, const float* resid, float* oF,
                  unsigned short* oU, int M, int N, int K, int mode) {
    gemm_bf16_kernel<<<dim3(N / 128, M / 128), dim3(256), 0, stream>>>(
        A, WT, bias, resid, oF, oU, M, N, K, mode);
  };

  // 1) weights -> bf16, transposed to [N][K] (column-contiguous B operands)
  tcvt(wq, wqT, Dm, Dm);   tcvt(wk, wkT, Dm, Dm);
  tcvt(wv, wvT, Dm, Dm);   tcvt(wo, woT, Dm, Dm);
  tcvt(w1, w1T, Dm, Dff);  tcvt(w2, w2T, Dff, Dm);

  // 2) LN1 -> bf16 activations
  ln_kernel<<<dim3((unsigned)BS), dim3(256), 0, stream>>>(hidden, ln1_g, ln1_b, xln, Dm);

  // 3) QKV projections (bf16 WMMA GEMMs, TDM-fed); V written transposed
  gemm(xln, wqT, bq, nullptr, nullptr, qbuf, (int)BS, Dm, Dm, 0);
  gemm(xln, wkT, bk, nullptr, nullptr, kbuf, (int)BS, Dm, Dm, 0);
  gemm(xln, wvT, bv, nullptr, nullptr, vTb,  (int)BS, Dm, Dm, 3);

  // 4) RoPE on q,k
  {
    long total = BS * H * 32;
    rope_kernel<<<dim3((unsigned)((total + 255) / 256)), dim3(256), 0, stream>>>(qbuf, kbuf, BS, S);
  }

  // 5) banded + global-token attention
  attn_kernel<<<dim3(S / 16, Bb * H), dim3(32), 0, stream>>>(
      qbuf, kbuf, vTb, gidx, n_glob, ctx, S, H, (int)BS);

  // 6) output projection + residual -> h (fp32)
  gemm(ctx, woT, bo, hidden, hbuf, nullptr, (int)BS, Dm, Dm, 1);

  // 7) LN2 -> bf16
  ln_kernel<<<dim3((unsigned)BS), dim3(256), 0, stream>>>(hbuf, ln2_g, ln2_b, ybuf, Dm);

  // 8) MLP up + GELU (bf16)
  gemm(ybuf, w1T, b1, nullptr, nullptr, act, (int)BS, Dff, Dm, 2);

  // 9) MLP down + residual -> d_out (fp32)
  gemm(act, w2T, b2, hbuf, out, nullptr, (int)BS, Dm, Dff, 1);

  (void)n_in; (void)out_size; (void)ws_size;
}
